// MultiheadAttention_18365280157699
// MI455X (gfx1250) — matrix operand order repaired and checked by reading, not compiled, not run on hardware
//
#include <hip/hip_runtime.h>

typedef __attribute__((ext_vector_type(16))) __bf16 v16bf;
typedef __attribute__((ext_vector_type(8)))  __bf16 v8bf;
typedef __attribute__((ext_vector_type(8)))  float  v8f;
typedef __attribute__((ext_vector_type(4)))  int    v4i;

#define D_MODEL 1024
#define LSEQ    2048
#define NHEAD   16
#define DK      64
#define BATCH   4
#define BL      (BATCH * LSEQ)   // 8192 rows total

// ---------------- WMMA helpers (CDNA5 gfx1250, wave32) ----------------

static __device__ __forceinline__ v8f wmma_bf16(v16bf a, v16bf b, v8f c) {
  // 8 args: (neg_a, A, neg_b, B, c_mod, C, reuse_a, reuse_b)
  return __builtin_amdgcn_wmma_f32_16x16x32_bf16(false, a, false, b, (short)0, c,
                                                 false, false);
}

// A-fragment 16x32 bf16: lane 0-15 -> M=lane, K chunks [0..7],[16..23];
// lane 16-31 -> M=lane-16, K chunks [8..15],[24..31]. Row-major source, K contiguous.
static __device__ __forceinline__ v16bf frag_a(const __bf16* base, int ld, int m0,
                                               int k0, int lane) {
  int m = lane & 15, g = lane >> 4;
  const __bf16* p = base + (size_t)(m0 + m) * ld + k0 + g * 8;
  v8bf lo = *(const v8bf*)(p);
  v8bf hi = *(const v8bf*)(p + 16);
  v16bf a;
#pragma unroll
  for (int i = 0; i < 8; i++) { a[i] = lo[i]; a[i + 8] = hi[i]; }
  return a;
}

// B-fragment 32x16 bf16: lane 0-15 -> N=lane, K chunks [0..7],[16..23]; lane 16-31 -> N=lane-16, K chunks [8..15],[24..31].
// Source is [N, K] row-major (torch-Linear weight layout), K contiguous.
static __device__ __forceinline__ v16bf frag_b(const __bf16* base, int ld, int n0,
                                               int k0, int lane) {
  int n = lane & 15, g = lane >> 4;
  const __bf16* p = base + (size_t)(n0 + n) * ld + k0 + g * 8;
  v8bf lo = *(const v8bf*)(p);
  v8bf hi = *(const v8bf*)(p + 16);
  v16bf b;
#pragma unroll
  for (int i = 0; i < 8; i++) { b[i] = lo[i]; b[i + 8] = hi[i]; }
  return b;
}

// reductions within 16-lane halves of a wave32 (xor masks < 16 never cross halves)
static __device__ __forceinline__ float rmax16(float v) {
#pragma unroll
  for (int m = 1; m < 16; m <<= 1) v = fmaxf(v, __shfl_xor(v, m, 32));
  return v;
}
static __device__ __forceinline__ float rsum16(float v) {
#pragma unroll
  for (int m = 1; m < 16; m <<= 1) v += __shfl_xor(v, m, 32);
  return v;
}

// ---------------- CDNA5 async global->LDS copy (ASYNCcnt-tracked) ----------------
// Builtin prototype (from compiler diagnostic): (v4i AS1* src, v4i AS3* dst, Ii, Ii).

typedef __attribute__((address_space(1))) v4i* g4p;
typedef __attribute__((address_space(3))) v4i* l4p;

static __device__ __forceinline__ void async_ld16(const void* g, void* l) {
#if __has_builtin(__builtin_amdgcn_global_load_async_to_lds_b128)
  __builtin_amdgcn_global_load_async_to_lds_b128((g4p)(size_t)g,
                                                 (l4p)(unsigned)(size_t)l, 0, 0);
#else
  unsigned lofs = (unsigned)(size_t)l;   // low 32 bits of generic shared ptr = LDS offset
  asm volatile("global_load_async_to_lds_b128 %0, %1, off"
               :: "v"(lofs), "v"(g) : "memory");
#endif
}

static __device__ __forceinline__ void wait_async0() {
#if __has_builtin(__builtin_amdgcn_s_wait_asynccnt)
  __builtin_amdgcn_s_wait_asynccnt(0);
#else
  asm volatile("s_wait_asynccnt 0" ::: "memory");
#endif
}

// ---------------- kernel 1: fp32 -> bf16 convert ----------------

__global__ void cvt_kernel(const float* __restrict__ in, __bf16* __restrict__ out, int n) {
  int i = blockIdx.x * blockDim.x + threadIdx.x;
  if (i < n) out[i] = (__bf16)in[i];
}

// ---------------- kernel 2: projection GEMM  out = (A @ W^T + bias) * scale ------
// block: 256 thr = 8 waves as 2 m-tiles x 4 n-chunks; wave tile = 16 x 128.
// W panel (512 cols x 32 k, 32 KB) is double-buffered in LDS via async loads:
// iteration s:  s_wait_asynccnt 0 ; barrier ; issue panel s+1 ; WMMA from panel s.
// LDS B-fragment reads are pipelined one tile ahead; A fragment double-buffered.

__global__ void proj_gemm_kernel(const __bf16* __restrict__ A, const __bf16* __restrict__ W,
                                 const float* __restrict__ bias, __bf16* __restrict__ outN,
                                 __bf16* __restrict__ outT, float outScale, int transposed) {
  __shared__ __align__(16) __bf16 wpanel[2][512][32];   // 2 x 32 KB
  int tid = threadIdx.x;
  int lane = tid & 31, wid = tid >> 5;
  int g = lane >> 4, ln = lane & 15;
  int mw = wid & 1, nw = wid >> 1;
  int m0 = blockIdx.x * 32 + mw * 16;
  int n0blk = blockIdx.y * 512;
  int n0 = n0blk + nw * 128;

  v8f acc[8];
#pragma unroll
  for (int t = 0; t < 8; t++) acc[t] = (v8f){0.f, 0.f, 0.f, 0.f, 0.f, 0.f, 0.f, 0.f};

  // issue panel for k=0 into buffer 0 (8 x 16B chunks per thread = 32 KB / block)
#pragma unroll
  for (int j = 0; j < 8; j++) {
    int c = j * 256 + tid;
    int nl = c >> 2, q = c & 3;
    async_ld16(W + (size_t)(n0blk + nl) * D_MODEL + q * 8, &wpanel[0][nl][q * 8]);
  }

  v16bf a = frag_a(A, D_MODEL, m0, 0, lane);   // A fragment for k-step 0

  for (int s = 0; s < D_MODEL / 32; ++s) {
    int k0 = s * 32;
    wait_async0();        // my panel-s chunks have landed
    __syncthreads();      // everyone's have landed; everyone done reading panel s+1's buffer
    if (s + 1 < D_MODEL / 32) {
      int kn = k0 + 32, nb = (s + 1) & 1;
#pragma unroll
      for (int j = 0; j < 8; j++) {
        int c = j * 256 + tid;
        int nl = c >> 2, q = c & 3;
        async_ld16(W + (size_t)(n0blk + nl) * D_MODEL + kn + q * 8, &wpanel[nb][nl][q * 8]);
      }
    }
    int cb = s & 1;
    // pipelined LDS reads: issue fragment t+1 before consuming fragment t
    v16bf bcur = frag_b(&wpanel[cb][0][0], 32, nw * 128, 0, lane);
    v16bf an;
#pragma unroll
    for (int t = 0; t < 8; t++) {
      v16bf bnext;
      if (t + 1 < 8) bnext = frag_b(&wpanel[cb][0][0], 32, nw * 128 + (t + 1) * 16, 0, lane);
      if (t == 0 && s + 1 < D_MODEL / 32)    // prefetch next A during the WMMA burst
        an = frag_a(A, D_MODEL, m0, k0 + 32, lane);
      acc[t] = wmma_bf16(a, bcur, acc[t]);
      bcur = bnext;
    }
    a = an;
  }

#pragma unroll
  for (int t = 0; t < 8; t++) {
    int col = n0 + t * 16 + ln;
    float bb = bias[col];
#pragma unroll
    for (int r = 0; r < 8; r++) {
      int row = m0 + r + 8 * g;
      float v = (acc[t][r] + bb) * outScale;
      if (!transposed) {
        outN[(size_t)row * D_MODEL + col] = (__bf16)v;
      } else {
        int bidx = row >> 11, l = row & (LSEQ - 1);
        outT[(size_t)bidx * ((size_t)D_MODEL * LSEQ) + (size_t)col * LSEQ + l] = (__bf16)v;
      }
    }
  }
}

// ---------------- kernel 3: flash attention forward (context + m/l) ----------------
// 1 wave = 16 q rows of one (b,h); 8 waves/block; grid = B*NHEAD*(L/128).
// mask+V loads issued first (in-order VMEM => S-WMMAs don't wait on them);
// next k-step's K fragments prefetched during the softmax VALU phase.

__global__ void attn_fwd_kernel(const __bf16* __restrict__ Qp, const __bf16* __restrict__ Kp,
                                const __bf16* __restrict__ VpT,
                                const unsigned char* __restrict__ mask,
                                float* __restrict__ ctx_f32, __bf16* __restrict__ ctx_bf,
                                float* __restrict__ Msc, float* __restrict__ Lsc) {
  __shared__ __align__(16) __bf16 pbuf[8][16][32];  // per-wave P staging (C->A relayout)
  int lane = threadIdx.x & 31, wid = threadIdx.x >> 5;
  int g = lane >> 4, ln = lane & 15;
  int bid = blockIdx.x;
  int b = bid >> 8;
  int h = (bid >> 4) & 15;
  int qg = bid & 15;
  int q0 = qg * 128 + wid * 16;
  int hc = h * DK;
  size_t bL = (size_t)b * LSEQ;

  v16bf aq0 = frag_a(Qp, D_MODEL, (int)(bL + q0), hc, lane);       // scale pre-folded in Qp
  v16bf aq1 = frag_a(Qp, D_MODEL, (int)(bL + q0), hc + 32, lane);

  v8f acc0 = {}, acc1 = {}, acc2 = {}, acc3 = {};
  float m[8], l[8];
#pragma unroll
  for (int r = 0; r < 8; r++) { m[r] = -3.0e38f; l[r] = 0.f; }

  const __bf16* vtb = VpT + (size_t)b * ((size_t)D_MODEL * LSEQ);

  v16bf kb0 = frag_b(Kp, D_MODEL, (int)(bL + 0), hc, lane);
  v16bf kb1 = frag_b(Kp, D_MODEL, (int)(bL + 0), hc + 32, lane);
  v16bf kb2 = frag_b(Kp, D_MODEL, (int)(bL + 16), hc, lane);
  v16bf kb3 = frag_b(Kp, D_MODEL, (int)(bL + 16), hc + 32, lane);

  for (int k0 = 0; k0 < LSEQ; k0 += 32) {
    // early loads: mask bytes + V fragments (complete during softmax phase)
    unsigned int mk0[8], mk1[8];
#pragma unroll
    for (int r = 0; r < 8; r++) {
      int q = q0 + r + 8 * g;
      const unsigned char* mp = mask + (bL + q) * (size_t)LSEQ;
      mk0[r] = mp[k0 + ln];
      mk1[r] = mp[k0 + 16 + ln];
    }
    v16bf bv0 = frag_b(vtb, LSEQ, hc + 0,  k0, lane);
    v16bf bv1 = frag_b(vtb, LSEQ, hc + 16, k0, lane);
    v16bf bv2 = frag_b(vtb, LSEQ, hc + 32, k0, lane);
    v16bf bv3 = frag_b(vtb, LSEQ, hc + 48, k0, lane);

    v8f s0 = {}, s1 = {};
    s0 = wmma_bf16(aq0, kb0, s0);
    s0 = wmma_bf16(aq1, kb1, s0);
    s1 = wmma_bf16(aq0, kb2, s1);
    s1 = wmma_bf16(aq1, kb3, s1);

    // prefetch next iteration's K fragments
    if (k0 + 32 < LSEQ) {
      kb0 = frag_b(Kp, D_MODEL, (int)(bL + k0 + 32), hc, lane);
      kb1 = frag_b(Kp, D_MODEL, (int)(bL + k0 + 32), hc + 32, lane);
      kb2 = frag_b(Kp, D_MODEL, (int)(bL + k0 + 48), hc, lane);
      kb3 = frag_b(Kp, D_MODEL, (int)(bL + k0 + 48), hc + 32, lane);
    }

#pragma unroll
    for (int r = 0; r < 8; r++) {   // mask: scores -> -1e9 where mask true
      if (mk0[r]) s0[r] = -1e9f;
      if (mk1[r]) s1[r] = -1e9f;
    }
#pragma unroll
    for (int r = 0; r < 8; r++) {   // online softmax update per row
      float v = rmax16(fmaxf(s0[r], s1[r]));
      float mn = fmaxf(m[r], v);
      float sc = __expf(m[r] - mn);
      float p0 = __expf(s0[r] - mn);
      float p1 = __expf(s1[r] - mn);
      s0[r] = p0; s1[r] = p1;
      l[r] = l[r] * sc + rsum16(p0 + p1);
      m[r] = mn;
      acc0[r] *= sc; acc1[r] *= sc; acc2[r] *= sc; acc3[r] *= sc;
    }
#pragma unroll
    for (int r = 0; r < 8; r++) {   // P (C layout) -> LDS, row-major 16x32
      int row = r + 8 * g;
      pbuf[wid][row][ln]      = (__bf16)s0[r];
      pbuf[wid][row][16 + ln] = (__bf16)s1[r];
    }
    __asm__ volatile("s_wait_dscnt 0" ::: "memory");  // per-wave LDS store->load fence
    v16bf ap;
    {
      const __bf16* pr = &pbuf[wid][ln][0];   // reload with A-fragment pattern
      v8bf lo = *(const v8bf*)(pr + g * 8);
      v8bf hi = *(const v8bf*)(pr + 16 + g * 8);
#pragma unroll
      for (int i = 0; i < 8; i++) { ap[i] = lo[i]; ap[i + 8] = hi[i]; }
    }
    acc0 = wmma_bf16(ap, bv0, acc0);
    acc1 = wmma_bf16(ap, bv1, acc1);
    acc2 = wmma_bf16(ap, bv2, acc2);
    acc3 = wmma_bf16(ap, bv3, acc3);
    __asm__ volatile("" ::: "memory");  // keep next iteration's LDS stores behind loads
  }

#pragma unroll
  for (int r = 0; r < 8; r++) {
    float inv = 1.0f / l[r];
    int q = q0 + r + 8 * g;
    size_t orow = (bL + q) * (size_t)D_MODEL + hc;
    float v0 = acc0[r] * inv, v1 = acc1[r] * inv, v2 = acc2[r] * inv, v3 = acc3[r] * inv;
    ctx_f32[orow + ln]      = v0;  ctx_bf[orow + ln]      = (__bf16)v0;
    ctx_f32[orow + 16 + ln] = v1;  ctx_bf[orow + 16 + ln] = (__bf16)v1;
    ctx_f32[orow + 32 + ln] = v2;  ctx_bf[orow + 32 + ln] = (__bf16)v2;
    ctx_f32[orow + 48 + ln] = v3;  ctx_bf[orow + 48 + ln] = (__bf16)v3;
  }
  if (ln == 0) {
#pragma unroll
    for (int r = 0; r < 8; r++) {
      int q = q0 + r + 8 * g;
      size_t idx = (size_t)(b * NHEAD + h) * LSEQ + q;
      Msc[idx] = m[r];
      Lsc[idx] = l[r];
    }
  }
}

// ---------------- kernel 4: attn_mean = mean_h softmax(S) ----------------
// 1 wave = one 16x16 (q,k) tile of one batch; recompute S per head via WMMA,
// with next head's fragments prefetched during the exp/accumulate phase.

__global__ void attn_mean_kernel(const __bf16* __restrict__ Qp, const __bf16* __restrict__ Kp,
                                 const unsigned char* __restrict__ mask,
                                 const float* __restrict__ Msc, const float* __restrict__ Lsc,
                                 float* __restrict__ am) {
  int lane = threadIdx.x & 31, wid = threadIdx.x >> 5;
  int g = lane >> 4, ln = lane & 15;
  int w = blockIdx.x * 8 + wid;           // 0 .. 65535
  int kt = w & 127;
  int qt = (w >> 7) & 127;
  int b  = w >> 14;
  int q0 = qt * 16, k0 = kt * 16;
  size_t bL = (size_t)b * LSEQ;

  v16bf a0 = frag_a(Qp, D_MODEL, (int)(bL + q0), 0, lane);
  v16bf a1 = frag_a(Qp, D_MODEL, (int)(bL + q0), 32, lane);
  v16bf b0 = frag_b(Kp, D_MODEL, (int)(bL + k0), 0, lane);
  v16bf b1 = frag_b(Kp, D_MODEL, (int)(bL + k0), 32, lane);

  v8f accv = {};
  for (int h = 0; h < NHEAD; h++) {
    v8f s = {};
    s = wmma_bf16(a0, b0, s);
    s = wmma_bf16(a1, b1, s);
    if (h + 1 < NHEAD) {           // prefetch next head's fragments
      int hc = (h + 1) * DK;
      a0 = frag_a(Qp, D_MODEL, (int)(bL + q0), hc, lane);
      a1 = frag_a(Qp, D_MODEL, (int)(bL + q0), hc + 32, lane);
      b0 = frag_b(Kp, D_MODEL, (int)(bL + k0), hc, lane);
      b1 = frag_b(Kp, D_MODEL, (int)(bL + k0), hc + 32, lane);
    }
    const float* mrow = Msc + (size_t)(b * NHEAD + h) * LSEQ;
    const float* lrow = Lsc + (size_t)(b * NHEAD + h) * LSEQ;
#pragma unroll
    for (int r = 0; r < 8; r++) {
      int q = q0 + r + 8 * g;
      float mh = mrow[q];
      float il = 1.0f / lrow[q];
      unsigned char mk = mask[(bL + q) * (size_t)LSEQ + k0 + ln];
      accv[r] += mk ? 0.0f : __expf(s[r] - mh) * il;
    }
  }
#pragma unroll
  for (int r = 0; r < 8; r++) {
    int q = q0 + r + 8 * g;
    am[(bL + q) * (size_t)LSEQ + k0 + ln] = accv[r] * (1.0f / NHEAD);
  }
}

// ---------------- kernel 5: O-proj + residual + LayerNorm, fused ----------------
// block = 16 rows x full 1024 cols; 8 waves x 128 cols each.
// register double buffering: next k-step's A/B fragments load while current WMMAs run.

__global__ void out_ln_kernel(const __bf16* __restrict__ Ctxb, const __bf16* __restrict__ Wob,
                              const float* __restrict__ bo, const float* __restrict__ resid,
                              const float* __restrict__ gamma, const float* __restrict__ beta,
                              float* __restrict__ outp) {
  __shared__ float ssum[16], ssq[16];
  int lane = threadIdx.x & 31, wid = threadIdx.x >> 5;
  int g = lane >> 4, ln = lane & 15;
  int m0 = blockIdx.x * 16;
  int n0 = wid * 128;

  if (threadIdx.x < 16) { ssum[threadIdx.x] = 0.f; ssq[threadIdx.x] = 0.f; }
  __syncthreads();

  v8f acc[8];
#pragma unroll
  for (int t = 0; t < 8; t++) acc[t] = (v8f){0.f, 0.f, 0.f, 0.f, 0.f, 0.f, 0.f, 0.f};

  v16bf a = frag_a(Ctxb, D_MODEL, m0, 0, lane);
  v16bf bw[8];
#pragma unroll
  for (int t = 0; t < 8; t++) bw[t] = frag_b(Wob, D_MODEL, n0 + t * 16, 0, lane);

  for (int k0 = 0; k0 < D_MODEL; k0 += 32) {
    v16bf an;
    v16bf bwn[8];
    if (k0 + 32 < D_MODEL) {     // prefetch next k-step
      an = frag_a(Ctxb, D_MODEL, m0, k0 + 32, lane);
#pragma unroll
      for (int t = 0; t < 8; t++) bwn[t] = frag_b(Wob, D_MODEL, n0 + t * 16, k0 + 32, lane);
    }
#pragma unroll
    for (int t = 0; t < 8; t++) acc[t] = wmma_bf16(a, bw[t], acc[t]);
    a = an;
#pragma unroll
    for (int t = 0; t < 8; t++) bw[t] = bwn[t];
  }

  float psum[8], psq[8];
#pragma unroll
  for (int r = 0; r < 8; r++) { psum[r] = 0.f; psq[r] = 0.f; }
#pragma unroll
  for (int t = 0; t < 8; t++) {
    int col = n0 + t * 16 + ln;
    float bb = bo[col];
#pragma unroll
    for (int r = 0; r < 8; r++) {
      int row = m0 + r + 8 * g;
      float v = acc[t][r] + bb + resid[(size_t)row * D_MODEL + col];
      acc[t][r] = v;
      psum[r] += v;
      psq[r] += v * v;
    }
  }
#pragma unroll
  for (int r = 0; r < 8; r++) {
    float s = rsum16(psum[r]);
    float s2 = rsum16(psq[r]);
    if (ln == 0) {
      atomicAdd(&ssum[r + 8 * g], s);
      atomicAdd(&ssq[r + 8 * g], s2);
    }
  }
  __syncthreads();
#pragma unroll
  for (int t = 0; t < 8; t++) {
    int col = n0 + t * 16 + ln;
    float gm = gamma[col], bt = beta[col];
#pragma unroll
    for (int r = 0; r < 8; r++) {
      int rr = r + 8 * g;
      float mu = ssum[rr] * (1.0f / D_MODEL);
      float var = ssq[rr] * (1.0f / D_MODEL) - mu * mu;
      float rstd = rsqrtf(var + 1e-5f);
      outp[(size_t)(m0 + rr) * D_MODEL + col] = (acc[t][r] - mu) * rstd * gm + bt;
    }
  }
}

// ---------------- launcher ----------------

extern "C" void kernel_launch(void* const* d_in, const int* in_sizes, int n_in,
                              void* d_out, int out_size, void* d_ws, size_t ws_size,
                              hipStream_t stream) {
  const float* Q = (const float*)d_in[0];
  const float* K = (const float*)d_in[1];
  const float* V = (const float*)d_in[2];
  const unsigned char* mask = (const unsigned char*)d_in[3];
  const float* Wq = (const float*)d_in[4];
  const float* bq = (const float*)d_in[5];
  const float* Wk = (const float*)d_in[6];
  const float* bk = (const float*)d_in[7];
  const float* Wv = (const float*)d_in[8];
  const float* bv = (const float*)d_in[9];
  const float* Wo = (const float*)d_in[10];
  const float* bo = (const float*)d_in[11];
  const float* gamma = (const float*)d_in[12];
  const float* beta  = (const float*)d_in[13];

  const size_t NE_X = (size_t)BL * D_MODEL;        // 8388608 elems
  const size_t NE_W = (size_t)D_MODEL * D_MODEL;   // 1048576 elems

  __bf16* p   = (__bf16*)d_ws;
  __bf16* Qb  = p;
  __bf16* Kb  = Qb + NE_X;
  __bf16* Vb  = Kb + NE_X;
  __bf16* Wqb = Vb + NE_X;
  __bf16* Wkb = Wqb + NE_W;
  __bf16* Wvb = Wkb + NE_W;
  __bf16* Wob = Wvb + NE_W;
  __bf16* Qp  = Wob + NE_W;
  __bf16* Kp  = Qp + NE_X;
  __bf16* VpT = Kp + NE_X;
  __bf16* Ctxb = VpT + NE_X;
  float*  Msc = (float*)(Ctxb + NE_X);
  float*  Lsc = Msc + (size_t)BATCH * NHEAD * LSEQ;

  float* outp = (float*)d_out;       // [B,L,D] output
  float* ctxO = outp + NE_X;         // [B,L,D] context
  float* amO  = ctxO + NE_X;         // [B,L,L] attn_mean

  dim3 blk(256);
  cvt_kernel<<<(int)(NE_X / 256), blk, 0, stream>>>(Q, Qb, (int)NE_X);
  cvt_kernel<<<(int)(NE_X / 256), blk, 0, stream>>>(K, Kb, (int)NE_X);
  cvt_kernel<<<(int)(NE_X / 256), blk, 0, stream>>>(V, Vb, (int)NE_X);
  cvt_kernel<<<(int)(NE_W / 256), blk, 0, stream>>>(Wq, Wqb, (int)NE_W);
  cvt_kernel<<<(int)(NE_W / 256), blk, 0, stream>>>(Wk, Wkb, (int)NE_W);
  cvt_kernel<<<(int)(NE_W / 256), blk, 0, stream>>>(Wv, Wvb, (int)NE_W);
  cvt_kernel<<<(int)(NE_W / 256), blk, 0, stream>>>(Wo, Wob, (int)NE_W);

  dim3 gproj(BL / 32, D_MODEL / 512);
  proj_gemm_kernel<<<gproj, blk, 0, stream>>>(Qb, Wqb, bq, Qp, nullptr, 0.125f, 0);  // 1/sqrt(64) folded
  proj_gemm_kernel<<<gproj, blk, 0, stream>>>(Kb, Wkb, bk, Kp, nullptr, 1.0f, 0);
  proj_gemm_kernel<<<gproj, blk, 0, stream>>>(Vb, Wvb, bv, nullptr, VpT, 1.0f, 1);

  attn_fwd_kernel<<<BATCH * NHEAD * (LSEQ / 128), blk, 0, stream>>>(
      Qp, Kp, VpT, mask, ctxO, Ctxb, Msc, Lsc);
  attn_mean_kernel<<<(BATCH * (LSEQ / 16) * (LSEQ / 16)) / 8, blk, 0, stream>>>(
      Qp, Kp, mask, Msc, Lsc, amO);
  out_ln_kernel<<<BL / 16, blk, 0, stream>>>(Ctxb, Wob, bo, Q, gamma, beta, outp);
}
